// EdgeNetwork_57226144252663
// MI455X (gfx1250) — compile-verified, matrix operand order checked by reading, not run on hardware
//
#include <hip/hip_runtime.h>
#include <hip/hip_bf16.h>
#include <math.h>
#include <stdint.h>

typedef __attribute__((ext_vector_type(16))) _Float16 v16h;
typedef __attribute__((ext_vector_type(8)))  _Float16 v8h;
typedef __attribute__((ext_vector_type(8)))  float    v8f;
typedef __attribute__((ext_vector_type(4)))  float    v4f;

namespace {
constexpr int Bb = 8;
constexpr int Nn = 50000;
constexpr int Ee = 200000;
constexpr int Dd = 16;
constexpr float EPSV = 1e-5f;
constexpr int TPB = Ee / 16;        // 12500 edge-tiles per batch (E % 16 == 0)
constexpr int NTILES = Bb * TPB;    // 100000 tiles of 16 edges
constexpr int ACT_STRIDE = 136;     // 128 + 8 f16 pad -> conflict-free b128 LDS loads
constexpr int WAVES = 8;            // 256 threads = 8 wave32
constexpr int NBLOCKS = 1250;       // 10000 waves, 10 tiles/wave
}

// sum across a 16-lane half (rows of one C fragment live in one half)
#define REDUCE_HALF16(x) do { \
  (x) += __shfl_xor((x), 1, 32); \
  (x) += __shfl_xor((x), 2, 32); \
  (x) += __shfl_xor((x), 4, 32); \
  (x) += __shfl_xor((x), 8, 32); } while (0)

// hardware transcendental tanh (v_tanh_f32 on gfx1250), 2-trans (exp2 + rcp) fallback otherwise
__device__ __forceinline__ float fast_tanh(float x) {
#if __has_builtin(__builtin_amdgcn_tanhf)
  return __builtin_amdgcn_tanhf(x);
#else
  float e = __builtin_amdgcn_exp2f(x * 2.8853900817779268f);  // exp(2x)
  return 1.f - 2.f * __builtin_amdgcn_rcpf(e + 1.f);
#endif
}

__device__ __forceinline__ float fast_sigmoid(float z) {
  float e = __builtin_amdgcn_exp2f(-z * 1.4426950408889634f);  // exp(-z)
  return __builtin_amdgcn_rcpf(1.f + e);
}

// ---- one-time: W^T in f16, row-major over the input (K) dimension ----
__global__ void prep_weights_kernel(const float* __restrict__ W1,
                                    const float* __restrict__ W2,
                                    const float* __restrict__ W3,
                                    _Float16* __restrict__ w1t,
                                    _Float16* __restrict__ w2t,
                                    _Float16* __restrict__ w3t) {
  int i = blockIdx.x * blockDim.x + threadIdx.x;   // i = k*128 + n
  if (i < 32 * 128) { int k = i >> 7, n = i & 127; w1t[n * 32 + k] = (_Float16)W1[i]; }
  if (i < 128 * 128) {
    int k = i >> 7, n = i & 127;
    w2t[n * 128 + k] = (_Float16)W2[i];
    w3t[n * 128 + k] = (_Float16)W3[i];
  }
}

__device__ __forceinline__ v16h cat8(v8h a, v8h b) {
  v16h r;
#pragma unroll
  for (int i = 0; i < 8; ++i) { r[i] = a[i]; r[8 + i] = b[i]; }
  return r;
}

// per-row mean / rsqrt(var) for the 16x128 result held as 8 C fragments
__device__ __forceinline__ void row_stats(const v8f c[8], float mu[8], float rs[8]) {
  float sum[8], sq[8];
#pragma unroll
  for (int r = 0; r < 8; ++r) { sum[r] = 0.f; sq[r] = 0.f; }
#pragma unroll
  for (int t = 0; t < 8; ++t)
#pragma unroll
    for (int r = 0; r < 8; ++r) { float v = c[t][r]; sum[r] += v; sq[r] = fmaf(v, v, sq[r]); }
#pragma unroll
  for (int r = 0; r < 8; ++r) { REDUCE_HALF16(sum[r]); REDUCE_HALF16(sq[r]); }
#pragma unroll
  for (int r = 0; r < 8; ++r) {
    mu[r] = sum[r] * (1.f / 128.f);
    float var = fmaf(sq[r], 1.f / 128.f, -mu[r] * mu[r]);
    rs[r] = __builtin_amdgcn_rsqf(var + EPSV);
  }
}

// layernorm + tanh, write 16x128 f16 activation tile to the wave's LDS buffer
__device__ __forceinline__ void ln_tanh_to_lds(const v8f c[8], const float (*sP)[128],
                                               int gIdx, int bIdx,
                                               int lo, int h, _Float16* act) {
  float mu[8], rs[8];
  row_stats(c, mu, rs);
#pragma unroll
  for (int t = 0; t < 8; ++t) {
    const int n = 16 * t + lo;
    const float ga = sP[gIdx][n];
    const float be = sP[bIdx][n];
#pragma unroll
    for (int r = 0; r < 8; ++r) {
      float a_ = rs[r] * ga;                 // scale
      float b_ = fmaf(-mu[r], a_, be);       // shift
      float y  = fmaf(c[t][r], a_, b_);      // one FMA per element
      act[(r + 8 * h) * ACT_STRIDE + n] = (_Float16)fast_tanh(y);
    }
  }
}

// [16x128] x [128x128] with bias-initialized accumulators; A from LDS, B from W^T
__device__ __forceinline__ void dense128(v8f c[8], const _Float16* wt,
                                         const float (*sP)[128], int biasIdx,
                                         int lo, int h, const _Float16* act) {
  v16h a[4];
#pragma unroll
  for (int k = 0; k < 4; ++k)   // K chunks of 32; layout matches 16-bit A fragment
    a[k] = cat8(*(const v8h*)(act + lo * ACT_STRIDE + 32 * k + 8 * h),
                *(const v8h*)(act + lo * ACT_STRIDE + 32 * k + 16 + 8 * h));
#pragma unroll
  for (int t = 0; t < 8; ++t) {
    const int n = 16 * t + lo;
    const float bias = sP[biasIdx][n];
    v8f acc;
#pragma unroll
    for (int r = 0; r < 8; ++r) acc[r] = bias;
#pragma unroll
    for (int k = 0; k < 4; ++k) {
      v16h bf = cat8(*(const v8h*)(wt + n * 128 + 32 * k + 8 * h),
                     *(const v8h*)(wt + n * 128 + 32 * k + 16 + 8 * h));
      acc = __builtin_amdgcn_wmma_f32_16x16x32_f16(false, a[k], false, bf,
                                                   (short)0, acc, false, false);
    }
    c[t] = acc;
  }
}

__global__ __launch_bounds__(256, 1)
void edge_mlp_kernel(const float* __restrict__ nodes,
                     const long long* __restrict__ edges,
                     const _Float16* __restrict__ w1t,
                     const _Float16* __restrict__ w2t,
                     const _Float16* __restrict__ w3t,
                     const float* __restrict__ b1, const float* __restrict__ g1, const float* __restrict__ be1,
                     const float* __restrict__ b2, const float* __restrict__ g2, const float* __restrict__ be2,
                     const float* __restrict__ b3, const float* __restrict__ g3, const float* __restrict__ be3,
                     const float* __restrict__ W4, const float* __restrict__ b4,
                     float* __restrict__ out) {
  __shared__ _Float16 sAct[WAVES][16][ACT_STRIDE];  // per-wave activation staging
  __shared__ float sP[10][128];                     // b1,g1,be1,b2,g2,be2,b3,g3,be3,w4

  if (threadIdx.x < 128) {
    int i = threadIdx.x;
    sP[0][i] = b1[i]; sP[1][i] = g1[i]; sP[2][i] = be1[i];
    sP[3][i] = b2[i]; sP[4][i] = g2[i]; sP[5][i] = be2[i];
    sP[6][i] = b3[i]; sP[7][i] = g3[i]; sP[8][i] = be3[i];
    sP[9][i] = W4[i];
  }
  __syncthreads();

  const int lane = threadIdx.x & 31;
  const int wave = threadIdx.x >> 5;
  const int lo = lane & 15;     // row (A) or column (B/C) within the 16-wide tile
  const int h  = lane >> 4;     // lane-half selects the K sub-block / row group
  const int gw = blockIdx.x * WAVES + wave;
  const int nw = gridDim.x * WAVES;
  const float b4s = b4[0];
  _Float16* act = &sAct[wave][0][0];

  uint64_t w1u = (uint64_t)w1t, w2u = (uint64_t)w2t, w3u = (uint64_t)w3t;

  for (int tile = gw; tile < NTILES; tile += nw) {
    // LICM blocker: keep the (loop-invariant) weight-fragment loads inside the
    // loop body so the compiler re-reads them from L2-hot global instead of
    // hoisting all 72 fragments and spilling them to scratch.
    asm volatile("" : "+s"(w1u), "+s"(w2u), "+s"(w3u));
    const _Float16* w1p = (const _Float16*)w1u;
    const _Float16* w2p = (const _Float16*)w2u;
    const _Float16* w3p = (const _Float16*)w3u;

    const int bb = tile / TPB;
    const int e0 = (tile - bb * TPB) * 16;

    // ---- gather: lane lo owns edge row lo; loads land directly in A-fragment layout.
    // concat-X K-map: K0..15 = src feats, K16..31 = dst feats; half h takes feats 8h..8h+7.
    const long long* ep = edges + (((long long)bb) * Ee + e0 + lo) * 2;
    const long long si = ep[0];
    const long long di = ep[1];
    const float* nb = nodes + ((long long)bb) * (Nn * Dd);   // L2-resident table
    const float* ps = nb + si * Dd + 8 * h;
    const float* pd = nb + di * Dd + 8 * h;
    v4f f0 = *(const v4f*)(ps), f1 = *(const v4f*)(ps + 4);
    v4f f2 = *(const v4f*)(pd), f3 = *(const v4f*)(pd + 4);
    v16h a1;
#pragma unroll
    for (int i = 0; i < 4; ++i) {
      a1[i]      = (_Float16)f0[i];
      a1[4 + i]  = (_Float16)f1[i];
      a1[8 + i]  = (_Float16)f2[i];
      a1[12 + i] = (_Float16)f3[i];
    }

    v8f c[8];

    // ---- layer 1: [16x32] x [32x128], one WMMA per 16-wide N tile ----
#pragma unroll
    for (int t = 0; t < 8; ++t) {
      const int n = 16 * t + lo;
      v16h bf = cat8(*(const v8h*)(w1p + n * 32 + 8 * h),
                     *(const v8h*)(w1p + n * 32 + 16 + 8 * h));
      const float bias = sP[0][n];
      v8f acc;
#pragma unroll
      for (int r = 0; r < 8; ++r) acc[r] = bias;
      c[t] = __builtin_amdgcn_wmma_f32_16x16x32_f16(false, a1, false, bf,
                                                    (short)0, acc, false, false);
    }
    ln_tanh_to_lds(c, sP, 1, 2, lo, h, act);

    // ---- layers 2 and 3: [16x128] x [128x128] (4 K-chunks x 8 N tiles each) ----
    dense128(c, w2p, sP, 3, lo, h, act);
    ln_tanh_to_lds(c, sP, 4, 5, lo, h, act);

    dense128(c, w3p, sP, 6, lo, h, act);

    // ---- layer-3 LN + tanh fused with the 128->1 dot, then sigmoid ----
    {
      float mu[8], rs[8];
      row_stats(c, mu, rs);
      float accd[8];
#pragma unroll
      for (int r = 0; r < 8; ++r) accd[r] = 0.f;
#pragma unroll
      for (int t = 0; t < 8; ++t) {
        const int n = 16 * t + lo;
        const float ga = sP[7][n];
        const float be = sP[8][n];
        const float w4 = sP[9][n];
#pragma unroll
        for (int r = 0; r < 8; ++r) {
          float a_ = rs[r] * ga;
          float b_ = fmaf(-mu[r], a_, be);
          float y  = fmaf(c[t][r], a_, b_);
          accd[r] = fmaf(fast_tanh(y), w4, accd[r]);
        }
      }
#pragma unroll
      for (int r = 0; r < 8; ++r) REDUCE_HALF16(accd[r]);
      if (lo == 0) {   // lanes 0 / 16 write rows 0-7 / 8-15
        float* op = out + ((long long)bb) * Ee + e0 + 8 * h;
#pragma unroll
        for (int r = 0; r < 8; ++r) op[r] = fast_sigmoid(accd[r] + b4s);
      }
    }
  }
}

extern "C" void kernel_launch(void* const* d_in, const int* in_sizes, int n_in,
                              void* d_out, int out_size, void* d_ws, size_t ws_size,
                              hipStream_t stream) {
  (void)in_sizes; (void)n_in; (void)out_size; (void)ws_size;
  const float*     nodes = (const float*)d_in[0];
  const long long* edges = (const long long*)d_in[1];
  const float* W1  = (const float*)d_in[2];
  const float* b1  = (const float*)d_in[3];
  const float* g1  = (const float*)d_in[4];
  const float* be1 = (const float*)d_in[5];
  const float* W2  = (const float*)d_in[6];
  const float* b2  = (const float*)d_in[7];
  const float* g2  = (const float*)d_in[8];
  const float* be2 = (const float*)d_in[9];
  const float* W3  = (const float*)d_in[10];
  const float* b3  = (const float*)d_in[11];
  const float* g3  = (const float*)d_in[12];
  const float* be3 = (const float*)d_in[13];
  const float* W4  = (const float*)d_in[14];
  const float* b4  = (const float*)d_in[15];

  _Float16* w1t = (_Float16*)d_ws;                            //  8 KB
  _Float16* w2t = (_Float16*)((char*)d_ws + 8192);            // 32 KB
  _Float16* w3t = (_Float16*)((char*)d_ws + 8192 + 32768);    // 32 KB

  prep_weights_kernel<<<64, 256, 0, stream>>>(W1, W2, W3, w1t, w2t, w3t);
  edge_mlp_kernel<<<NBLOCKS, 256, 0, stream>>>(
      nodes, edges, w1t, w2t, w3t,
      b1, g1, be1, b2, g2, be2, b3, g3, be3, W4, b4,
      (float*)d_out);
}
